// MosaicGNN_9405978378813
// MI455X (gfx1250) — compile-verified, matrix operand order checked by reading, not run on hardware
//
#include <hip/hip_runtime.h>

typedef __attribute__((ext_vector_type(16))) __bf16 v16bf;
typedef __attribute__((ext_vector_type(8)))  float  v8f;

// Native conversion: lets the backend emit hardware bf16 cvt ops
// (packed v_cvt_pk_bf16_f32 when vectorizable) instead of 3-op bit math.
__device__ __forceinline__ __bf16 f2bf(float f) { return (__bf16)f; }

// ---------------------------------------------------------------- zero fill
__global__ __launch_bounds__(256) void gnn_zero(float* __restrict__ p, long long n) {
  long long i = (long long)blockIdx.x * blockDim.x + threadIdx.x;
  long long stride = (long long)gridDim.x * blockDim.x;
  for (; i < n; i += stride) p[i] = 0.0f;
}

// ------------------------------------------------- edge scatter (mean agg)
// 16 lanes per edge, float4 per lane -> coalesced L2-resident atomics.
__global__ __launch_bounds__(256) void gnn_scatter(
    const float* __restrict__ x, const int* __restrict__ esrc,
    const int* __restrict__ edst, float* __restrict__ agg,
    float* __restrict__ cnt, int E, int docnt) {
  long long t = (long long)blockIdx.x * blockDim.x + threadIdx.x;
  long long e = t >> 4;
  int q = (int)(t & 15);
  if (e >= E) return;
  int s = esrc[e], d = edst[e];
  const float4 v = *(const float4*)(x + (long long)s * 64 + q * 4);
  float* o = agg + (long long)d * 64 + q * 4;
  atomicAdd(o + 0, v.x);
  atomicAdd(o + 1, v.y);
  atomicAdd(o + 2, v.z);
  atomicAdd(o + 3, v.w);
  if (docnt && q == 0) atomicAdd(cnt + d, 1.0f);
}

// ----------------------------------------------- projection GEMM (K=128)
// out[N,64] = x[N,128] @ w[128,64] + bias ; wave handles a 16x64 tile.
__global__ __launch_bounds__(256) void gnn_proj(
    const float* __restrict__ x, const float* __restrict__ w,
    const float* __restrict__ bias, float* __restrict__ out, int N) {
  __shared__ __bf16 Wt[64 * 128];   // Bt[n][k], bf16, 16KB
  __shared__ float sb[64];
  int tid = threadIdx.x;
  for (int i = tid; i < 64 * 128; i += 256) {
    int n = i >> 7, k = i & 127;
    Wt[i] = f2bf(w[k * 64 + n]);
  }
  if (tid < 64) sb[tid] = bias[tid];
  __syncthreads();
  int lane = tid & 31, wave = tid >> 5;
  long long tile = (long long)blockIdx.x * 8 + wave;
  long long row0 = tile * 16;
  if (row0 >= N) return;
  int row = lane & 15;
  int hi  = lane >> 4;
  int kb  = hi << 3;
  long long myrow = row0 + row; if (myrow > (long long)N - 1) myrow = (long long)N - 1;
  const float* pa = x + myrow * 128;
  v8f c[4] = {};
  #pragma unroll
  for (int ks = 0; ks < 4; ++ks) {
    int k0 = ks * 32 + kb;
    v16bf a;
    #pragma unroll
    for (int i = 0; i < 8; ++i) { a[i] = f2bf(pa[k0 + i]); a[i + 8] = f2bf(pa[k0 + 16 + i]); }
    #pragma unroll
    for (int nt = 0; nt < 4; ++nt) {
      const __bf16* q = Wt + (nt * 16 + row) * 128 + k0;
      v16bf b;
      #pragma unroll
      for (int i = 0; i < 8; ++i) { b[i] = q[i]; b[i + 8] = q[16 + i]; }
      c[nt] = __builtin_amdgcn_wmma_f32_16x16x32_bf16(false, a, false, b, (short)0, c[nt], false, false);
    }
  }
  #pragma unroll
  for (int nt = 0; nt < 4; ++nt) {
    int n = nt * 16 + row;
    float bv = sb[n];
    #pragma unroll
    for (int r = 0; r < 8; ++r) {
      long long orow = row0 + r + hi * 8;
      if (orow < N) out[orow * 64 + n] = c[nt][r] + bv;
    }
  }
}

// --------------------------------- SAGE combine: (agg/cnt)@wl + bl + x@wr
__global__ __launch_bounds__(256) void gnn_combine(
    const float* __restrict__ agg, const float* __restrict__ cnt,
    const float* __restrict__ xdst,
    const float* __restrict__ wl, const float* __restrict__ bl,
    const float* __restrict__ wr,
    float* __restrict__ out, int N, int relu) {
  __shared__ __bf16 WtL[64 * 64];   // 8KB each
  __shared__ __bf16 WtR[64 * 64];
  __shared__ float sbl[64];
  int tid = threadIdx.x;
  for (int i = tid; i < 64 * 64; i += 256) {
    int n = i >> 6, k = i & 63;
    WtL[i] = f2bf(wl[k * 64 + n]);
    WtR[i] = f2bf(wr[k * 64 + n]);
  }
  if (tid < 64) sbl[tid] = bl[tid];
  __syncthreads();
  int lane = tid & 31, wave = tid >> 5;
  long long tile = (long long)blockIdx.x * 8 + wave;
  long long row0 = tile * 16;
  if (row0 >= N) return;
  int row = lane & 15;
  int hi  = lane >> 4;
  int kb  = hi << 3;
  long long myrow = row0 + row; if (myrow > (long long)N - 1) myrow = (long long)N - 1;
  float rinv = 1.0f / fmaxf(cnt[myrow], 1.0f);
  const float* pa = agg  + myrow * 64;
  const float* px = xdst + myrow * 64;
  v8f c[4] = {};
  #pragma unroll
  for (int ks = 0; ks < 2; ++ks) {
    int k0 = ks * 32 + kb;
    v16bf a, a2;
    #pragma unroll
    for (int i = 0; i < 8; ++i) {
      a [i]     = f2bf(pa[k0 + i] * rinv);
      a [i + 8] = f2bf(pa[k0 + 16 + i] * rinv);
      a2[i]     = f2bf(px[k0 + i]);
      a2[i + 8] = f2bf(px[k0 + 16 + i]);
    }
    #pragma unroll
    for (int nt = 0; nt < 4; ++nt) {
      const __bf16* ql = WtL + (nt * 16 + row) * 64 + k0;
      const __bf16* qr = WtR + (nt * 16 + row) * 64 + k0;
      v16bf bL, bR;
      #pragma unroll
      for (int i = 0; i < 8; ++i) {
        bL[i] = ql[i]; bL[i + 8] = ql[16 + i];
        bR[i] = qr[i]; bR[i + 8] = qr[16 + i];
      }
      c[nt] = __builtin_amdgcn_wmma_f32_16x16x32_bf16(false, a,  false, bL, (short)0, c[nt], false, false);
      c[nt] = __builtin_amdgcn_wmma_f32_16x16x32_bf16(false, a2, false, bR, (short)0, c[nt], false, false);
    }
  }
  #pragma unroll
  for (int nt = 0; nt < 4; ++nt) {
    int n = nt * 16 + row;
    float bv = sbl[n];
    #pragma unroll
    for (int r = 0; r < 8; ++r) {
      float v = c[nt][r] + bv;
      if (relu) v = fmaxf(v, 0.0f);
      long long orow = row0 + r + hi * 8;
      if (orow < N) out[orow * 64 + n] = v;
    }
  }
}

// ------------------------------------------------------------- decoder
// per row: h = relu(concat(u,m) @ w1 + b1) ; out = h . w2 + b2
__global__ __launch_bounds__(256) void gnn_decoder(
    const float* __restrict__ user2, const float* __restrict__ movie2,
    const int* __restrict__ lu, const int* __restrict__ lm,
    const float* __restrict__ w1, const float* __restrict__ b1,
    const float* __restrict__ w2, const float* __restrict__ b2,
    float* __restrict__ out, int L) {
  __shared__ __bf16 Wt[64 * 128];   // Bt[n][k] of dec_w1, 16KB
  __shared__ float sb1[64], sw2[64];
  int tid = threadIdx.x;
  for (int i = tid; i < 64 * 128; i += 256) {
    int n = i >> 7, k = i & 127;
    Wt[i] = f2bf(w1[k * 64 + n]);
  }
  if (tid < 64) { sb1[tid] = b1[tid]; sw2[tid] = w2[tid]; }
  __syncthreads();
  int lane = tid & 31, wave = tid >> 5;
  long long tile = (long long)blockIdx.x * 8 + wave;
  long long row0 = tile * 16;
  if (row0 >= L) return;
  int row = lane & 15;
  int hi  = lane >> 4;
  int kb  = hi << 3;
  long long myrow = row0 + row; if (myrow > (long long)L - 1) myrow = (long long)L - 1;
  const float* pu = user2  + (long long)lu[myrow] * 64;
  const float* pm = movie2 + (long long)lm[myrow] * 64;
  v8f c[4] = {};
  #pragma unroll
  for (int ks = 0; ks < 4; ++ks) {      // K = 128 over concat(u[64], m[64])
    int k0 = ks * 32 + kb;
    int k1 = k0 + 16;
    const float* s0 = (k0 < 64) ? (pu + k0) : (pm + (k0 - 64));
    const float* s1 = (k1 < 64) ? (pu + k1) : (pm + (k1 - 64));
    v16bf a;
    #pragma unroll
    for (int i = 0; i < 8; ++i) { a[i] = f2bf(s0[i]); a[i + 8] = f2bf(s1[i]); }
    #pragma unroll
    for (int nt = 0; nt < 4; ++nt) {
      const __bf16* q = Wt + (nt * 16 + row) * 128 + k0;
      v16bf b;
      #pragma unroll
      for (int i = 0; i < 8; ++i) { b[i] = q[i]; b[i + 8] = q[16 + i]; }
      c[nt] = __builtin_amdgcn_wmma_f32_16x16x32_bf16(false, a, false, b, (short)0, c[nt], false, false);
    }
  }
  // second layer: out[M] = sum_N relu(h[M][N]) * w2[N] + b2, reduced in-register
  float b2v = b2[0];
  #pragma unroll
  for (int r = 0; r < 8; ++r) {
    float s = 0.0f;
    #pragma unroll
    for (int nt = 0; nt < 4; ++nt) {
      int n = nt * 16 + row;
      float v = c[nt][r] + sb1[n];
      v = fmaxf(v, 0.0f);
      s += v * sw2[n];
    }
    s += __shfl_xor(s, 1);
    s += __shfl_xor(s, 2);
    s += __shfl_xor(s, 4);
    s += __shfl_xor(s, 8);   // half-wave (16-lane) reduction; M = r + 8*hi
    long long orow = row0 + r + hi * 8;
    if (row == 0 && orow < L) out[orow] = s + b2v;
  }
}

extern "C" void kernel_launch(void* const* d_in, const int* in_sizes, int n_in,
                              void* d_out, int out_size, void* d_ws, size_t ws_size,
                              hipStream_t stream) {
  const float* movie_x  = (const float*)d_in[0];
  const int*   e_su     = (const int*)d_in[1];
  const int*   e_dm     = (const int*)d_in[2];
  const int*   lu       = (const int*)d_in[3];
  const int*   lm       = (const int*)d_in[4];
  const float* user_emb = (const float*)d_in[5];
  const float* proj_w   = (const float*)d_in[6];
  const float* proj_b   = (const float*)d_in[7];
  const float* dec_w1   = (const float*)d_in[8];
  const float* dec_b1   = (const float*)d_in[9];
  const float* dec_w2   = (const float*)d_in[10];
  const float* dec_b2   = (const float*)d_in[11];
  const float* w1_um_l  = (const float*)d_in[12];
  const float* b1_um_l  = (const float*)d_in[13];
  const float* w1_um_r  = (const float*)d_in[14];
  const float* w1_mu_l  = (const float*)d_in[15];
  const float* b1_mu_l  = (const float*)d_in[16];
  const float* w1_mu_r  = (const float*)d_in[17];
  const float* w2_um_l  = (const float*)d_in[18];
  const float* b2_um_l  = (const float*)d_in[19];
  const float* w2_um_r  = (const float*)d_in[20];
  const float* w2_mu_l  = (const float*)d_in[21];
  const float* b2_mu_l  = (const float*)d_in[22];
  const float* w2_mu_r  = (const float*)d_in[23];

  const int NM = in_sizes[0] / 128;
  const int E  = in_sizes[1];
  const int L  = in_sizes[3];
  const int NU = in_sizes[5] / 64;

  float* p = (float*)d_ws;
  float* movie0 = p; p += (size_t)NM * 64;
  float* movie1 = p; p += (size_t)NM * 64;
  float* movie2 = p; p += (size_t)NM * 64;
  float* user1  = p; p += (size_t)NU * 64;
  float* user2  = p; p += (size_t)NU * 64;
  float* agg    = p; p += (size_t)NU * 64;   // shared scratch, NU rows (movie passes assume NU >= NM)
  float* cntU   = p; p += NU;
  float* cntM   = p; p += NM;

  dim3 blk(256);
  long long sc_threads = (long long)E * 16;
  int sc_blocks = (int)((sc_threads + 255) / 256);
  int tilesM = (NM + 15) / 16, tilesU = (NU + 15) / 16, tilesL = (L + 15) / 16;

  // movie0 = movie_x @ proj_w + proj_b
  gnn_proj<<<(tilesM + 7) / 8, blk, 0, stream>>>(movie_x, proj_w, proj_b, movie0, NM);

  // conv1: user -> movie
  gnn_zero<<<1024, blk, 0, stream>>>(agg, (long long)NM * 64);
  gnn_zero<<<64,   blk, 0, stream>>>(cntM, NM);
  gnn_scatter<<<sc_blocks, blk, 0, stream>>>(user_emb, e_su, e_dm, agg, cntM, E, 1);
  gnn_combine<<<(tilesM + 7) / 8, blk, 0, stream>>>(agg, cntM, movie0, w1_um_l, b1_um_l, w1_um_r, movie1, NM, 1);

  // conv1: movie -> user
  gnn_zero<<<1024, blk, 0, stream>>>(agg, (long long)NU * 64);
  gnn_zero<<<64,   blk, 0, stream>>>(cntU, NU);
  gnn_scatter<<<sc_blocks, blk, 0, stream>>>(movie0, e_dm, e_su, agg, cntU, E, 1);
  gnn_combine<<<(tilesU + 7) / 8, blk, 0, stream>>>(agg, cntU, user_emb, w1_mu_l, b1_mu_l, w1_mu_r, user1, NU, 1);

  // conv2: user -> movie (counts reused)
  gnn_zero<<<1024, blk, 0, stream>>>(agg, (long long)NM * 64);
  gnn_scatter<<<sc_blocks, blk, 0, stream>>>(user1, e_su, e_dm, agg, cntM, E, 0);
  gnn_combine<<<(tilesM + 7) / 8, blk, 0, stream>>>(agg, cntM, movie1, w2_um_l, b2_um_l, w2_um_r, movie2, NM, 0);

  // conv2: movie -> user
  gnn_zero<<<1024, blk, 0, stream>>>(agg, (long long)NU * 64);
  gnn_scatter<<<sc_blocks, blk, 0, stream>>>(movie1, e_dm, e_su, agg, cntU, E, 0);
  gnn_combine<<<(tilesU + 7) / 8, blk, 0, stream>>>(agg, cntU, user1, w2_mu_l, b2_mu_l, w2_mu_r, user2, NU, 0);

  // decoder -> d_out [L]
  gnn_decoder<<<(tilesL + 7) / 8, blk, 0, stream>>>(user2, movie2, lu, lm,
                                                    dec_w1, dec_b1, dec_w2, dec_b2,
                                                    (float*)d_out, L);
  (void)n_in; (void)out_size; (void)ws_size;
}
